// BiRNNImputerModel_57088705298694
// MI455X (gfx1250) — matrix operand order repaired and checked by reading, not compiled, not run on hardware
//
#include <hip/hip_runtime.h>
#include <hip/hip_bf16.h>

// Problem constants (match reference)
#define BB 128
#define SS 512
#define FF 64
#define HH 512
// 2F = 128, 3H = 1536

typedef __attribute__((ext_vector_type(16))) __bf16 v16bf;
typedef __attribute__((ext_vector_type(8)))  float  v8f;

union FragU {
    struct { uint4 lo, hi; } q;
    v16bf v;
};

__device__ __forceinline__ unsigned short f2bf(float f) {
    unsigned int u = __float_as_uint(f);
    u += 0x7FFFu + ((u >> 16) & 1u);   // round-to-nearest-even
    return (unsigned short)(u >> 16);
}
__device__ __forceinline__ float bf2f(unsigned short h) {
    return __uint_as_float(((unsigned int)h) << 16);
}
__device__ __forceinline__ v8f splat8(float x) {
    v8f v;
#pragma unroll
    for (int i = 0; i < 8; ++i) v[i] = x;
    return v;
}
__device__ __forceinline__ float sigm(float x) {
    return 1.0f / (1.0f + __expf(-x));
}

// A fragment (16x32 bf16, MxK): row = base pointer of row m (= lane&15),
// lanes 0-15 (hf=0): halves[0..15]=K kb+0..15
// lanes 16-31 (hf=1): halves[0..15]=K kb+16..31
__device__ __forceinline__ v16bf loadA(const unsigned short* row, int kb, int hf) {
    FragU f;
    f.q.lo = *(const uint4*)(row + kb + 16 * hf);
    f.q.hi = *(const uint4*)(row + kb + 16 * hf + 8);
    return f.v;
}
// B fragment (32x16 bf16, KxN): p points at W[j][kb + 16*hf] (k-contiguous row),
// lane holds 16 consecutive K values for its column n = lane&15.
__device__ __forceinline__ v16bf loadB(const unsigned short* p) {
    FragU f;
    f.q.lo = *(const uint4*)(p);
    f.q.hi = *(const uint4*)(p + 8);
    return f.v;
}

#define WMMA_BF16(A, B, C) \
    __builtin_amdgcn_wmma_f32_16x16x32_bf16(false, (A), false, (B), (short)0, (C), false, false)

// ---------------------------------------------------------------------------
// fp32 -> bf16 weight conversion
// ---------------------------------------------------------------------------
__global__ void cvt_f32_bf16(const float* __restrict__ src,
                             unsigned short* __restrict__ dst, int n) {
    int i = blockIdx.x * blockDim.x + threadIdx.x;
    int stride = gridDim.x * blockDim.x;
    for (; i < n; i += stride) dst[i] = f2bf(src[i]);
}

// ---------------------------------------------------------------------------
// Persistent GRU kernel: one block per (batch tile of 16, direction).
// 1024 threads = 32 waves (fills one WGP). Wave w owns gate columns
// [w*16, w*16+16) of H=512 for all three gates (r, z, n):
//   - 4 live v8f accumulators per wave (32 VGPRs) -> no spills
//   - gate math entirely in registers
// Recurrent state h lives in LDS (bf16); weights stream from L2.
// Gate rows r/z/n are addressed from ONE base pointer per weight matrix via
// compile-time-constant offsets (fold into the 24-bit instruction IOFFSET).
// ---------------------------------------------------------------------------
__global__ __launch_bounds__(1024) void gru_kernel(
    const float* __restrict__ X, const unsigned char* __restrict__ Mk,
    const unsigned short* __restrict__ WihF, const unsigned short* __restrict__ WhhF,
    const unsigned short* __restrict__ WroF,
    const unsigned short* __restrict__ WihB, const unsigned short* __restrict__ WhhB,
    const unsigned short* __restrict__ WroB,
    const float* __restrict__ bihF, const float* __restrict__ bhhF, const float* __restrict__ broF,
    const float* __restrict__ bihB, const float* __restrict__ bhhB, const float* __restrict__ broB,
    unsigned short* __restrict__ hbufF, unsigned short* __restrict__ hbufB,
    float* __restrict__ xhF_out, float* __restrict__ xhB_out)
{
    const int dir = blockIdx.y;
    const int b0  = blockIdx.x * 16;

    const unsigned short* Wih = dir ? WihB : WihF;
    const unsigned short* Whh = dir ? WhhB : WhhF;
    const unsigned short* Wro = dir ? WroB : WroF;
    const float* bih = dir ? bihB : bihF;
    const float* bhh = dir ? bhhB : bhhF;
    const float* bro = dir ? broB : broF;
    unsigned short* hbuf = dir ? hbufB : hbufF;
    float* xh_out = dir ? xhB_out : xhF_out;

    __shared__ __align__(16) unsigned short h_lds[16][520];    // bf16 recurrent state
    __shared__ __align__(16) unsigned short xin_lds[16][136];  // bf16 [x_p | mask]
    __shared__ float xhat_lds[16][64];

    const int tid  = threadIdx.x;
    const int lane = tid & 31;
    const int w    = tid >> 5;     // wave 0..31
    const int nl   = lane & 15;
    const int hf   = lane >> 4;

    // ---- init: h = 0, xhat = bro; emit t=0 outputs ----
    for (int i = tid; i < 16 * 520; i += 1024) ((unsigned short*)h_lds)[i] = 0;
    if (tid < 16 * 64) {
        int m = tid >> 6, j = tid & 63;
        xhat_lds[m][j] = bro[j];
        int pos0 = dir ? (SS - 1) : 0;
        xh_out[((size_t)(b0 + m) * SS + pos0) * FF + j] = bro[j];
    }
    {
        int pos0 = dir ? (SS - 1) : 0;
        for (int i = tid; i < 16 * 512; i += 1024) {
            int m = i >> 9, j = i & 511;
            hbuf[((size_t)(b0 + m) * SS + pos0) * HH + j] = 0;
        }
    }
    __syncthreads();

    // per-lane loop-invariant constants: this lane's gate column
    const int jc = w * 16 + nl;                 // column within H owned by lane
    float brv  = bih[jc] + bhh[jc];
    float bzv  = bih[512 + jc] + bhh[512 + jc];
    float bniv = bih[1024 + jc];
    float bnhv = bhh[1024 + jc];
    const float broj = (w < 4) ? bro[w * 16 + nl] : 0.0f;

    // ONE base pointer per weight matrix; gate rows via constant offsets:
    //   Wih row stride 128:  z = +512*128 (65536),  n = +1024*128 (131072)
    //   Whh row stride 512:  z = +512*512 (262144), n = +1024*512 (524288)
    const unsigned short* wih0 = Wih + (size_t)jc * 128 + hf * 16;
    const unsigned short* whh0 = Whh + (size_t)jc * 512 + hf * 16;
    const unsigned short* wro_j = Wro + (size_t)(w * 16 + nl) * 512 + hf * 16;

    const unsigned short* arow_x = &xin_lds[nl][0];
    const unsigned short* arow_h = &h_lds[nl][0];

    for (int st = 0; st < SS - 1; ++st) {
        const int s   = dir ? (SS - 1 - st) : st;       // input time index consumed
        const int pos = dir ? (SS - 2 - st) : (st + 1); // output index of new h/xhat

        // Defeat LICM without destroying address-space inference:
        //  - opaque ZERO offset keeps Wih/Wro fragment loads inside the time
        //    loop (otherwise 12+ fragments get hoisted and spilled to scratch)
        //  - opaque bias scalars keep the 4 splat vectors from being hoisted
        //    and spilled (re-splatting is 8 v_mov; a scratch reload is worse)
        unsigned int zofs = 0;
        asm volatile("" : "+v"(zofs), "+v"(brv), "+v"(bzv), "+v"(bniv), "+v"(bnhv));
        const unsigned short* wi = wih0 + zofs;
        const unsigned short* wo = wro_j + zofs;

        // ---- build x_in = [m ? x : xhat , m] (bf16): one element per thread ----
        {
            int m = tid >> 6, k = tid & 63;
            size_t gi = ((size_t)(b0 + m) * SS + s) * FF + k;
            float xv = X[gi];
            bool mk = Mk[gi] != 0;
            xin_lds[m][k]      = f2bf(mk ? xv : xhat_lds[m][k]);
            xin_lds[m][64 + k] = f2bf(mk ? 1.0f : 0.0f);
        }
        __syncthreads();

        // ---- accumulators (fp32) initialized with biases ----
        v8f acc_r  = splat8(brv);
        v8f acc_z  = splat8(bzv);
        v8f acc_in = splat8(bniv);
        v8f acc_hn = splat8(bnhv);

        // ---- gi = x_in @ Wih.T  (K = 128) ----
        for (int kt = 0; kt < 4; ++kt) {
            const int kb = kt * 32;
            v16bf a = loadA(arow_x, kb, hf);
            acc_r  = WMMA_BF16(a, loadB(wi + kb), acc_r);
            acc_z  = WMMA_BF16(a, loadB(wi + kb + 512 * 128), acc_z);
            acc_in = WMMA_BF16(a, loadB(wi + kb + 1024 * 128), acc_in);
        }

        // ---- gh = h @ Whh.T  (K = 512) ----
        for (int kt = 0; kt < 16; ++kt) {
            const int kb = kt * 32;
            v16bf a = loadA(arow_h, kb, hf);
            acc_r  = WMMA_BF16(a, loadB(whh0 + kb), acc_r);
            acc_z  = WMMA_BF16(a, loadB(whh0 + kb + 512 * 512), acc_z);
            acc_hn = WMMA_BF16(a, loadB(whh0 + kb + 1024 * 512), acc_hn);
        }
        __syncthreads();   // all reads of h_lds complete

        // ---- gate math (in registers) + h update ----
#pragma unroll
        for (int q = 0; q < 8; ++q) {
            int m = q + 8 * hf;
            float r = sigm(acc_r[q]);
            float z = sigm(acc_z[q]);
            float n = tanhf(acc_in[q] + r * acc_hn[q]);
            float hold = bf2f(h_lds[m][jc]);
            float hnew = (1.0f - z) * n + z * hold;
            unsigned short hb = f2bf(hnew);
            h_lds[m][jc] = hb;
            hbuf[((size_t)(b0 + m) * SS + pos) * HH + jc] = hb;
        }
        __syncthreads();   // h_lds fully updated

        // ---- xhat = h_new @ Wro.T + bro  (waves 0..3, one 16x16 N-tile each) ----
        if (w < 4) {
            const int j = w * 16 + nl;
            v8f acc = splat8(broj);
            for (int kt = 0; kt < 16; ++kt) {
                const int kb = kt * 32;
                v16bf a = loadA(arow_h, kb, hf);
                v16bf b = loadB(wo + kb);
                acc = WMMA_BF16(a, b, acc);
            }
#pragma unroll
            for (int q = 0; q < 8; ++q) {
                int m = q + 8 * hf;
                xhat_lds[m][j] = acc[q];
                xh_out[((size_t)(b0 + m) * SS + pos) * FF + j] = acc[q];
            }
        }
        __syncthreads();
    }
}

// ---------------------------------------------------------------------------
// Final projection: x_hat[B*S, 64] = concat(h_f, h_b)[B*S, 1024] @ Wout.T + bout
// One block per M-tile of 16 rows; 4 waves = 4 N-tiles of 16; K = 1024.
// ---------------------------------------------------------------------------
__global__ __launch_bounds__(128) void out_gemm_kernel(
    const unsigned short* __restrict__ hF, const unsigned short* __restrict__ hB,
    const unsigned short* __restrict__ Wo, const float* __restrict__ bout,
    float* __restrict__ out)
{
    const int row0 = blockIdx.x * 16;      // over B*S
    const int lane = threadIdx.x & 31;
    const int w    = threadIdx.x >> 5;     // N tile 0..3
    const int nl   = lane & 15;
    const int hf   = lane >> 4;
    const int j    = w * 16 + nl;

    v8f acc = splat8(bout[j]);
    const unsigned short* aFrow = hF + (size_t)(row0 + nl) * HH;
    const unsigned short* aBrow = hB + (size_t)(row0 + nl) * HH;
    const unsigned short* brow  = Wo + (size_t)j * 1024 + hf * 16;

    for (int kt = 0; kt < 32; ++kt) {
        const int kb = kt * 32;
        const unsigned short* ar = (kb < 512) ? (aFrow + kb) : (aBrow + (kb - 512));
        v16bf a = loadA(ar, 0, hf);
        v16bf b = loadB(brow + kb);
        acc = WMMA_BF16(a, b, acc);
    }
#pragma unroll
    for (int q = 0; q < 8; ++q) {
        int m = q + 8 * hf;
        out[(size_t)(row0 + m) * FF + j] = acc[q];
    }
}

// ---------------------------------------------------------------------------
extern "C" void kernel_launch(void* const* d_in, const int* in_sizes, int n_in,
                              void* d_out, int out_size, void* d_ws, size_t ws_size,
                              hipStream_t stream) {
    (void)in_sizes; (void)n_in; (void)out_size; (void)ws_size;

    const float*         X     = (const float*)d_in[0];
    const unsigned char* Mk    = (const unsigned char*)d_in[1];
    const float* Wih_f = (const float*)d_in[2];
    const float* Whh_f = (const float*)d_in[3];
    const float* bih_f = (const float*)d_in[4];
    const float* bhh_f = (const float*)d_in[5];
    const float* Wro_f = (const float*)d_in[6];
    const float* bro_f = (const float*)d_in[7];
    const float* Wih_b = (const float*)d_in[8];
    const float* Whh_b = (const float*)d_in[9];
    const float* bih_b = (const float*)d_in[10];
    const float* bhh_b = (const float*)d_in[11];
    const float* Wro_b = (const float*)d_in[12];
    const float* bro_b = (const float*)d_in[13];
    const float* Wout  = (const float*)d_in[14];
    const float* bout  = (const float*)d_in[15];

    // workspace layout (bf16 elements)
    unsigned short* ws = (unsigned short*)d_ws;
    const size_t nH = (size_t)BB * SS * HH;           // 33,554,432
    unsigned short* hbufF = ws;
    unsigned short* hbufB = hbufF + nH;
    unsigned short* wihF  = hbufB + nH;
    unsigned short* whhF  = wihF + 3 * HH * 2 * FF;   // 196608
    unsigned short* wroF  = whhF + 3 * HH * HH;       // 786432
    unsigned short* wihB  = wroF + FF * HH;           // 32768
    unsigned short* whhB  = wihB + 3 * HH * 2 * FF;
    unsigned short* wroB  = whhB + 3 * HH * HH;
    unsigned short* woutW = wroB + FF * HH;

    auto cvt = [&](const float* s, unsigned short* d, int n) {
        int blocks = (n + 255) / 256;
        if (blocks > 1024) blocks = 1024;
        cvt_f32_bf16<<<blocks, 256, 0, stream>>>(s, d, n);
    };
    cvt(Wih_f, wihF, 3 * HH * 2 * FF);
    cvt(Whh_f, whhF, 3 * HH * HH);
    cvt(Wro_f, wroF, FF * HH);
    cvt(Wih_b, wihB, 3 * HH * 2 * FF);
    cvt(Whh_b, whhB, 3 * HH * HH);
    cvt(Wro_b, wroB, FF * HH);
    cvt(Wout,  woutW, FF * 2 * HH);

    float* x_hat = (float*)d_out;
    float* xh_f  = x_hat + (size_t)BB * SS * FF;
    float* xh_b  = xh_f + (size_t)BB * SS * FF;

    gru_kernel<<<dim3(BB / 16, 2), 1024, 0, stream>>>(
        X, Mk, wihF, whhF, wroF, wihB, whhB, wroB,
        bih_f, bhh_f, bro_f, bih_b, bhh_b, bro_b,
        hbufF, hbufB, xh_f, xh_b);

    out_gemm_kernel<<<dim3((BB * SS) / 16), 128, 0, stream>>>(
        hbufF, hbufB, woutW, bout, x_hat);
}
